// SwinTransformerBlock_94489281361
// MI455X (gfx1250) — matrix operand order repaired and checked by reading, not compiled, not run on hardware
//
#include <hip/hip_runtime.h>
#include <hip/hip_bf16.h>
#include <math.h>

typedef __bf16 bf16;
typedef __attribute__((ext_vector_type(16))) __bf16 v16bf;
typedef __attribute__((ext_vector_type(8)))  __bf16 v8bf;
typedef __attribute__((ext_vector_type(8)))  float  v8f;
typedef __attribute__((ext_vector_type(4)))  int    v4i;

#if defined(__has_builtin)
# if __has_builtin(__builtin_amdgcn_global_load_async_to_lds_b128)
#  define USE_ASYNC 1
# endif
# if __has_builtin(__builtin_amdgcn_s_wait_asynccnt)
#  define HAVE_WAIT_ASYNC_BUILTIN 1
# endif
#endif

#define DEVI __device__ __forceinline__

// ---------------------------------------------------------------------------
// async global->LDS 16B copy (CDNA5 GLOBAL_LOAD_ASYNC_TO_LDS_B128), with
// synchronous fallback if the builtin is unavailable on this toolchain.
// Builtin signature (ROCm clang): (AS1 v4i*, AS3 v4i*, imm offset, imm cpol)
// ---------------------------------------------------------------------------
DEVI void cp16_g2l(bf16* lds, const bf16* g) {
#if defined(USE_ASYNC)
  __builtin_amdgcn_global_load_async_to_lds_b128(
      (__attribute__((address_space(1))) v4i*)g,
      (__attribute__((address_space(3))) v4i*)lds, 0, 0);
#else
  *(v8bf*)lds = *(const v8bf*)g;
#endif
}
template <int N>
DEVI void wait_async() {
#if defined(USE_ASYNC)
# if defined(HAVE_WAIT_ASYNC_BUILTIN)
  __builtin_amdgcn_s_wait_asynccnt(N);
# else
  if constexpr (N == 0) asm volatile("s_wait_asynccnt 0x0" ::: "memory");
  else                  asm volatile("s_wait_asynccnt 0x2" ::: "memory");
# endif
#endif
}

// ---------------------------------------------------------------------------
// WMMA fragment loaders (CDNA5 bf16 16x16x32 lane layouts, wave32)
// A (16x32): lanes 0-15 -> M=lane, K=k0..k0+7 / k0+16..k0+23 (k0=0); lanes
//            16-31 -> M=lane-16, k0=8
// B (32x16): lanes 0-15 -> N=lane, K=0..7 / 16..23 ; lanes 16-31 -> N=lane-16, K=8..15 / 24..31
// C/D f32:   lane -> N=lane&15 ; vgpr r -> M = r + (lane&16 ? 8 : 0)
// ---------------------------------------------------------------------------
DEVI v16bf frag_a(const bf16* base, int ld, int lane) {
  const bf16* p = base + (size_t)(lane & 15) * ld + ((lane & 16) ? 8 : 0);
  union { v16bf v; v8bf h[2]; } u;
  u.h[0] = *(const v8bf*)(p);
  u.h[1] = *(const v8bf*)(p + 16);
  return u.v;
}
DEVI v16bf frag_b(const bf16* base, int ld, int lane) {
  const bf16* p = base + (size_t)(lane & 15) * ld + ((lane & 16) ? 8 : 0);
  union { v16bf v; v8bf h[2]; } u;
  u.h[0] = *(const v8bf*)(p);
  u.h[1] = *(const v8bf*)(p + 16);
  return u.v;
}
DEVI v8f wmma_bf16(v16bf a, v16bf b, v8f c) {
  return __builtin_amdgcn_wmma_f32_16x16x32_bf16(false, a, false, b, (short)0, c,
                                                 false, false);
}

// ---------------------------------------------------------------------------
__global__ __launch_bounds__(256) void cvt_k(const float* __restrict__ s,
                                             bf16* __restrict__ d, int n) {
  for (int i = blockIdx.x * 256 + threadIdx.x; i < n; i += gridDim.x * 256)
    d[i] = (bf16)s[i];
}

// CPB MLP: t[225][12]  +  per-head scale = exp(min(logit_scale, ln 100))
__global__ __launch_bounds__(256) void cpb_k(
    const float* __restrict__ table, const float* __restrict__ w1,
    const float* __restrict__ b1, const float* __restrict__ w2,
    const float* __restrict__ ls, float* __restrict__ tout,
    float* __restrict__ scale) {
  int i = threadIdx.x;
  if (i < 225) {
    float x0 = table[2 * i], x1 = table[2 * i + 1];
    float hid[64];
#pragma unroll
    for (int j = 0; j < 64; ++j)
      hid[j] = fmaxf(w1[2 * j] * x0 + w1[2 * j + 1] * x1 + b1[j], 0.f);
#pragma unroll
    for (int h = 0; h < 12; ++h) {
      float s = 0.f;
#pragma unroll
      for (int j = 0; j < 64; ++j) s += w2[h * 64 + j] * hid[j];
      tout[i * 12 + h] = s;
    }
  }
  if (i < 12) scale[i] = __expf(fminf(ls[i], 4.6051701859880914f));
}

// fused bias+mask: bm[h][wm][n][m] = 16*sigmoid(t[idx[n][m]][h]) + mask[wm][n][m]
__global__ __launch_bounds__(256) void bmfuse_k(const float* __restrict__ tout,
                                                const int* __restrict__ idx,
                                                const float* __restrict__ amask,
                                                float* __restrict__ bm) {
  int i = blockIdx.x * 256 + threadIdx.x;
  if (i < 12 * 64 * 4096) {
    int h = i >> 18, rem = i & 0x3FFFF;
    int wm = rem >> 12, nm = rem & 4095;
    float v = tout[idx[nm] * 12 + h];
    bm[i] = 16.f / (1.f + __expf(-v)) + amask[wm * 4096 + nm];
  }
}

// cyclic shift (-4,-4) + window partition + bf16 convert: x -> hb (65536 x 768)
__global__ __launch_bounds__(192) void shiftin_k(const float* __restrict__ x,
                                                 bf16* __restrict__ hb) {
  int tk = blockIdx.x;               // b*4096 + w*64 + n
  int b = tk >> 12, rem = tk & 4095;
  int w = rem >> 6, n = rem & 63;
  int wy = w >> 3, wx = w & 7, iy = n >> 3, ix = n & 7;
  int y = (wy * 8 + iy + 4) & 63, xp = (wx * 8 + ix + 4) & 63;
  const float4* src = (const float4*)(x + ((size_t)b * 4096 + y * 64 + xp) * 768);
  bf16* dst = hb + (size_t)tk * 768;
  int i = threadIdx.x;              // 192 threads * float4 = 768
  float4 v = src[i];
  dst[4 * i + 0] = (bf16)v.x; dst[4 * i + 1] = (bf16)v.y;
  dst[4 * i + 2] = (bf16)v.z; dst[4 * i + 3] = (bf16)v.w;
}

// ---------------------------------------------------------------------------
// bf16 WMMA GEMM: out[M x N] = A[M x K] * W[N x K]^T, templated epilogue.
// WG = 256 thr (8 waves), block tile 64(M) x 256(N), wave tile 32 x 64.
// A tile double-buffer staged in LDS via async global->LDS copies (ASYNCcnt);
// B (weights) read from global (L2-resident).
// EPI 0: qkv scatter->q/k/vt(bf16)  1: +bias, f32  2: GELU->bf16  3: f32
// ---------------------------------------------------------------------------
template <int EPI>
__global__ __launch_bounds__(256) void gemm_k(
    const bf16* __restrict__ A, const bf16* __restrict__ W, int N, int K,
    float* __restrict__ outF, bf16* __restrict__ outB,
    const float* __restrict__ bias, bf16* __restrict__ qO,
    bf16* __restrict__ kO, bf16* __restrict__ vtO) {
  __shared__ bf16 As[2][64 * 72];
  const int t = threadIdx.x;
  const int lane = t & 31, wid = t >> 5;
  const int bm0 = blockIdx.x * 64;
  const int bm = bm0 + (wid >> 2) * 32;
  const int bn = blockIdx.y * 256 + (wid & 3) * 64;
  const int crow = t >> 2, cch = (t & 3) * 8;       // copy duty: row + chunk
  const bf16* gROW = A + (size_t)(bm0 + crow) * K;
  v8f acc[2][4] = {};
  const int S = K >> 6;  // 64-wide K stages
  cp16_g2l(&As[0][crow * 72 + cch], gROW + cch);
  cp16_g2l(&As[0][crow * 72 + cch + 32], gROW + cch + 32);
  for (int s = 0; s < S; ++s) {
    if (s + 1 < S) {
      const int kn = (s + 1) << 6;
      bf16* dst = As[(s + 1) & 1];
      cp16_g2l(&dst[crow * 72 + cch], gROW + kn + cch);
      cp16_g2l(&dst[crow * 72 + cch + 32], gROW + kn + cch + 32);
      wait_async<2>();   // previous stage's copies done (in-order completion)
    } else {
      wait_async<0>();
    }
    __syncthreads();
    const bf16* Ab = As[s & 1] + ((wid >> 2) * 32) * 72;
#pragma unroll
    for (int kc = 0; kc < 64; kc += 32) {
      const int kk = (s << 6) + kc;
      v16bf a0 = frag_a(Ab + kc, 72, lane);
      v16bf a1 = frag_a(Ab + 16 * 72 + kc, 72, lane);
      v16bf b0 = frag_b(W + (size_t)bn * K + kk, K, lane);
      v16bf b1 = frag_b(W + (size_t)(bn + 16) * K + kk, K, lane);
      v16bf b2 = frag_b(W + (size_t)(bn + 32) * K + kk, K, lane);
      v16bf b3 = frag_b(W + (size_t)(bn + 48) * K + kk, K, lane);
      acc[0][0] = wmma_bf16(a0, b0, acc[0][0]);
      acc[0][1] = wmma_bf16(a0, b1, acc[0][1]);
      acc[0][2] = wmma_bf16(a0, b2, acc[0][2]);
      acc[0][3] = wmma_bf16(a0, b3, acc[0][3]);
      acc[1][0] = wmma_bf16(a1, b0, acc[1][0]);
      acc[1][1] = wmma_bf16(a1, b1, acc[1][1]);
      acc[1][2] = wmma_bf16(a1, b2, acc[1][2]);
      acc[1][3] = wmma_bf16(a1, b3, acc[1][3]);
    }
    __syncthreads();
  }
#pragma unroll
  for (int mt = 0; mt < 2; ++mt)
#pragma unroll
    for (int nt = 0; nt < 4; ++nt) {
      v8f c = acc[mt][nt];
      int col = bn + nt * 16 + (lane & 15);
#pragma unroll
      for (int r = 0; r < 8; ++r) {
        int row = bm + mt * 16 + r + ((lane & 16) ? 8 : 0);
        float v = c[r];
        if constexpr (EPI == 0) {  // qkv: col = which*768 + head*64 + d
          int w = row >> 6, ntk = row & 63;
          int which = col / 768, rr = col - which * 768;
          int head = rr >> 6, d = rr & 63;
          size_t hb_ = ((size_t)w * 12 + head) * 64;
          if (which == 0)      qO[(hb_ + ntk) * 64 + d] = (bf16)v;
          else if (which == 1) kO[(hb_ + ntk) * 64 + d] = (bf16)v;
          else                 vtO[(hb_ + d) * 64 + ntk] = (bf16)v;  // V^T
        } else if constexpr (EPI == 1) {
          outF[(size_t)row * N + col] = v + bias[col];
        } else if constexpr (EPI == 2) {
          float g = 0.5f * v * (1.0f + erff(v * 0.70710678118654752f));
          outB[(size_t)row * N + col] = (bf16)g;
        } else {
          outF[(size_t)row * N + col] = v;
        }
      }
    }
}

// ---------------------------------------------------------------------------
// attention per (window, head): cosine attn + fused bias/mask + softmax + P@V
// ---------------------------------------------------------------------------
__global__ __launch_bounds__(256) void attn_k(
    const bf16* __restrict__ Q, const bf16* __restrict__ Kb,
    const bf16* __restrict__ Vt, const float* __restrict__ bm,
    const float* __restrict__ scale, bf16* __restrict__ ao) {
  const int w = blockIdx.x, h = blockIdx.y;
  const int t = threadIdx.x;
  __shared__ bf16 qs[64 * 72];   // normalized q, later reused for P
  __shared__ bf16 ks[64 * 72];   // normalized k
  __shared__ float ss[64 * 68];  // scores
  const size_t base = ((size_t)w * 12 + h) * 4096;
  const bf16* qp = Q + base;
  const bf16* kp = Kb + base;
  if (t < 128) {  // one row per thread: L2 normalize
    const bf16* src = (t < 64) ? (qp + (size_t)t * 64) : (kp + (size_t)(t - 64) * 64);
    v8bf ch[8];
    float s = 0.f;
#pragma unroll
    for (int j = 0; j < 8; ++j) {
      ch[j] = *(const v8bf*)(src + j * 8);
#pragma unroll
      for (int e = 0; e < 8; ++e) { float f = (float)ch[j][e]; s += f * f; }
    }
    float inv = 1.0f / fmaxf(sqrtf(s), 1e-12f);
    bf16* dst = (t < 64) ? (qs + t * 72) : (ks + (t - 64) * 72);
#pragma unroll
    for (int j = 0; j < 8; ++j)
#pragma unroll
      for (int e = 0; e < 8; ++e) dst[j * 8 + e] = (bf16)((float)ch[j][e] * inv);
  }
  __syncthreads();
  const int lane = t & 31, wid = t >> 5;
  const float sc = scale[h];
  const float* bmh = bm + ((size_t)h * 64 + (w & 63)) * 4096;
#pragma unroll
  for (int ti = 0; ti < 2; ++ti) {  // 16 tiles over 8 waves
    int tile = wid * 2 + ti;
    int tm = tile >> 2, tn = tile & 3;
    v8f c = {};
#pragma unroll
    for (int kk = 0; kk < 64; kk += 32) {
      v16bf a = frag_a(qs + tm * 16 * 72 + kk, 72, lane);
      v16bf b = frag_b(ks + tn * 16 * 72 + kk, 72, lane);
      c = wmma_bf16(a, b, c);
    }
    int col = tn * 16 + (lane & 15);
#pragma unroll
    for (int r = 0; r < 8; ++r) {
      int row = tm * 16 + r + ((lane & 16) ? 8 : 0);
      ss[row * 68 + col] = c[r] * sc + bmh[row * 64 + col];
    }
  }
  __syncthreads();
  if (t < 64) {  // softmax over keys, write P (bf16) into qs
    float mx = -3.4e38f;
    for (int j = 0; j < 64; ++j) mx = fmaxf(mx, ss[t * 68 + j]);
    float sum = 0.f;
    for (int j = 0; j < 64; ++j) {
      float e = __expf(ss[t * 68 + j] - mx);
      ss[t * 68 + j] = e; sum += e;
    }
    float inv = 1.0f / sum;
    for (int j = 0; j < 64; ++j) qs[t * 72 + j] = (bf16)(ss[t * 68 + j] * inv);
  }
  __syncthreads();
  const bf16* vp = Vt + base;  // vt[d][m]
#pragma unroll
  for (int ti = 0; ti < 2; ++ti) {
    int tile = wid * 2 + ti;
    int tm = tile >> 2, tn = tile & 3;
    v8f c = {};
#pragma unroll
    for (int kk = 0; kk < 64; kk += 32) {
      v16bf a = frag_a(qs + tm * 16 * 72 + kk, 72, lane);
      v16bf b = frag_b(vp + (size_t)tn * 16 * 64 + kk, 64, lane);
      c = wmma_bf16(a, b, c);
    }
    int d = tn * 16 + (lane & 15);
#pragma unroll
    for (int r = 0; r < 8; ++r) {
      int n = tm * 16 + r + ((lane & 16) ? 8 : 0);
      ao[((size_t)w * 64 + n) * 768 + h * 64 + d] = (bf16)c[r];
    }
  }
}

// ---------------------------------------------------------------------------
// window reverse + unshift + LN + residual: x1 = x + LN(y);  x1b = bf16(x1)
// ---------------------------------------------------------------------------
__global__ __launch_bounds__(256) void ln1_k(
    const float* __restrict__ y, const float* __restrict__ x,
    const float* __restrict__ g, const float* __restrict__ b,
    float* __restrict__ x1, bf16* __restrict__ x1b) {
  int tok = blockIdx.x;
  int bb = tok >> 12, l = tok & 4095;
  int yy = l >> 6, xx = l & 63;
  int py = (yy - 4) & 63, px = (xx - 4) & 63;
  int w = (py >> 3) * 8 + (px >> 3), n = (py & 7) * 8 + (px & 7);
  const float* src = y + (((size_t)bb * 64 + w) * 64 + n) * 768;
  const float* xs = x + (size_t)tok * 768;
  __shared__ float s1[256], s2[256];
  int t = threadIdx.x;
  float a = 0.f, q = 0.f;
  for (int c = t; c < 768; c += 256) { float v = src[c]; a += v; q += v * v; }
  s1[t] = a; s2[t] = q; __syncthreads();
  for (int o = 128; o > 0; o >>= 1) {
    if (t < o) { s1[t] += s1[t + o]; s2[t] += s2[t + o]; }
    __syncthreads();
  }
  float mean = s1[0] * (1.f / 768.f);
  float var = s2[0] * (1.f / 768.f) - mean * mean;
  float rstd = rsqrtf(var + 1e-5f);
  for (int c = t; c < 768; c += 256) {
    float v = xs[c] + (src[c] - mean) * rstd * g[c] + b[c];
    x1[(size_t)tok * 768 + c] = v;
    x1b[(size_t)tok * 768 + c] = (bf16)v;
  }
}

// out += LN(m)
__global__ __launch_bounds__(256) void ln2_k(const float* __restrict__ m,
                                             const float* __restrict__ g,
                                             const float* __restrict__ b,
                                             float* __restrict__ io) {
  int tok = blockIdx.x;
  const float* src = m + (size_t)tok * 768;
  float* o = io + (size_t)tok * 768;
  __shared__ float s1[256], s2[256];
  int t = threadIdx.x;
  float a = 0.f, q = 0.f;
  for (int c = t; c < 768; c += 256) { float v = src[c]; a += v; q += v * v; }
  s1[t] = a; s2[t] = q; __syncthreads();
  for (int off = 128; off > 0; off >>= 1) {
    if (t < off) { s1[t] += s1[t + off]; s2[t] += s2[t + off]; }
    __syncthreads();
  }
  float mean = s1[0] * (1.f / 768.f);
  float var = s2[0] * (1.f / 768.f) - mean * mean;
  float rstd = rsqrtf(var + 1e-5f);
  for (int c = t; c < 768; c += 256)
    o[c] = o[c] + (src[c] - mean) * rstd * g[c] + b[c];
}

// ---------------------------------------------------------------------------
extern "C" void kernel_launch(void* const* d_in, const int* in_sizes, int n_in,
                              void* d_out, int out_size, void* d_ws,
                              size_t ws_size, hipStream_t stream) {
  (void)in_sizes; (void)n_in; (void)out_size; (void)ws_size;
  const float* x      = (const float*)d_in[0];
  const float* qkv_w  = (const float*)d_in[1];
  const float* proj_w = (const float*)d_in[2];
  const float* proj_b = (const float*)d_in[3];
  const float* lscale = (const float*)d_in[4];
  const float* cpb_w1 = (const float*)d_in[5];
  const float* cpb_b1 = (const float*)d_in[6];
  const float* cpb_w2 = (const float*)d_in[7];
  const float* n1w    = (const float*)d_in[8];
  const float* n1b    = (const float*)d_in[9];
  const float* n2w    = (const float*)d_in[10];
  const float* n2b    = (const float*)d_in[11];
  const float* mlp_w1 = (const float*)d_in[12];
  const float* mlp_w2 = (const float*)d_in[13];
  const float* table  = (const float*)d_in[14];
  const float* amask  = (const float*)d_in[15];
  const int*   rpi    = (const int*)d_in[16];
  float* out = (float*)d_out;

  char* p = (char*)d_ws;
  auto alloc = [&](size_t bytes) -> char* {
    char* r = p; p += (bytes + 255) & ~(size_t)255; return r;
  };
  const size_t NT = 65536;  // tokens
  bf16* qkv_wb = (bf16*)alloc((size_t)2304 * 768 * 2);
  bf16* proj_wb = (bf16*)alloc((size_t)768 * 768 * 2);
  bf16* w1b = (bf16*)alloc((size_t)3072 * 768 * 2);
  bf16* w2b = (bf16*)alloc((size_t)768 * 3072 * 2);
  float* tcpb = (float*)alloc(225 * 12 * 4);
  float* scale = (float*)alloc(12 * 4);
  float* bm = (float*)alloc((size_t)12 * 64 * 4096 * 4);  // fused bias+mask
  bf16* hb = (bf16*)alloc(NT * 768 * 2);   // shifted windows; reused as attn out
  bf16* qb = (bf16*)alloc(NT * 768 * 2);   // q; reused as x1 bf16
  bf16* kb = (bf16*)alloc(NT * 768 * 2);
  bf16* vtb = (bf16*)alloc(NT * 768 * 2);
  float* ybuf = (float*)alloc(NT * 768 * 4);  // proj out; reused as mlp2 out
  bf16* m1 = (bf16*)alloc(NT * 3072 * 2);     // gelu(mlp1)

  cvt_k<<<2048, 256, 0, stream>>>(qkv_w, qkv_wb, 2304 * 768);
  cvt_k<<<2048, 256, 0, stream>>>(proj_w, proj_wb, 768 * 768);
  cvt_k<<<2048, 256, 0, stream>>>(mlp_w1, w1b, 3072 * 768);
  cvt_k<<<2048, 256, 0, stream>>>(mlp_w2, w2b, 768 * 3072);
  cpb_k<<<1, 256, 0, stream>>>(table, cpb_w1, cpb_b1, cpb_w2, lscale, tcpb, scale);
  bmfuse_k<<<(12 * 64 * 4096 + 255) / 256, 256, 0, stream>>>(tcpb, rpi, amask, bm);
  shiftin_k<<<65536, 192, 0, stream>>>(x, hb);
  // qkv: (65536 x 768) * (2304 x 768)^T -> scatter q/k/vt
  gemm_k<0><<<dim3(1024, 9), 256, 0, stream>>>(hb, qkv_wb, 2304, 768, nullptr,
                                               nullptr, nullptr, qb, kb, vtb);
  attn_k<<<dim3(1024, 12), 256, 0, stream>>>(qb, kb, vtb, bm, scale, hb);
  // proj: (65536 x 768) * (768 x 768)^T + b -> ybuf (f32, window layout)
  gemm_k<1><<<dim3(1024, 3), 256, 0, stream>>>(hb, proj_wb, 768, 768, ybuf,
                                               nullptr, proj_b, nullptr, nullptr,
                                               nullptr);
  ln1_k<<<65536, 256, 0, stream>>>(ybuf, x, n1w, n1b, out, qb);
  // mlp1 + gelu: (65536 x 768) * (3072 x 768)^T -> m1 (bf16)
  gemm_k<2><<<dim3(1024, 12), 256, 0, stream>>>(qb, w1b, 3072, 768, nullptr, m1,
                                                nullptr, nullptr, nullptr,
                                                nullptr);
  // mlp2: (65536 x 3072) * (768 x 3072)^T -> ybuf (f32)
  gemm_k<3><<<dim3(1024, 3), 256, 0, stream>>>(m1, w2b, 768, 3072, ybuf, nullptr,
                                               nullptr, nullptr, nullptr,
                                               nullptr);
  ln2_k<<<65536, 256, 0, stream>>>(ybuf, n2w, n2b, out);
}
